// GatedExpertsEncoder_34969623724297
// MI455X (gfx1250) — matrix operand order repaired and checked by reading, not compiled, not run on hardware
//
#include <hip/hip_runtime.h>
#include <hip/hip_bf16.h>
#include <cstdint>
#include <cstddef>

// Problem constants (match reference)
#define B_ROWS 65536
#define DIM    512      // D_IN == D_H == D_OUT
#define NEXP   8
#define GH     128      // gating hidden / N_GATE

typedef __bf16 bf16_t;
typedef __attribute__((ext_vector_type(16))) __bf16 v16bf;
typedef __attribute__((ext_vector_type(8)))  float  v8f;

union Frag { uint4 u[2]; v16bf v; };

__device__ inline unsigned short bf_bits(float f) {
  union { float f; unsigned u; } v; v.f = f;
  unsigned r = v.u + 0x7FFFu + ((v.u >> 16) & 1u);   // round-to-nearest-even
  return (unsigned short)(r >> 16);
}
__device__ inline bf16_t f2bf(float f) {
  union { unsigned short s; bf16_t b; } o; o.s = bf_bits(f); return o.b;
}
__device__ inline float elu_f(float x) { return x > 0.0f ? x : expm1f(x); }

__device__ inline void store_val(float*  p, float v) { *p = v; }
__device__ inline void store_val(bf16_t* p, float v) { *p = f2bf(v); }

// ---- CDNA5 async global->LDS copy (ASYNCcnt-tracked, bypasses VGPRs) ------
// LDS byte address: flat->LDS mapping uses addr[31:0], so truncating the
// generic pointer yields the LDS offset the instruction expects.
__device__ inline void async_copy_b128(const void* lds_dst, const void* gsrc) {
  unsigned lds_off = (unsigned)(unsigned long long)lds_dst;
  asm volatile("global_load_async_to_lds_b128 %0, %1, off"
               :: "v"(lds_off), "v"(gsrc) : "memory");
}
__device__ inline void wait_async0() {
#if __has_builtin(__builtin_amdgcn_s_wait_asynccnt)
  __builtin_amdgcn_s_wait_asynccnt(0);
#else
  asm volatile("s_wait_asynccnt 0x0" ::: "memory");
#endif
}

// ---- CDNA5 Tensor Data Mover: 2D tile -> LDS (TENSORcnt-tracked DMA) ------
#if __has_builtin(__builtin_amdgcn_tensor_load_to_lds)
#define HAVE_TDM 1
typedef __attribute__((ext_vector_type(4))) unsigned tdm_v4u;
typedef __attribute__((ext_vector_type(8))) int      tdm_v8i;
typedef __attribute__((ext_vector_type(4))) int      tdm_v4i;

// D# per CDNA5 ISA (groups 0/1; groups 2/3 zero for a 2D tensor).
// Element size 4B (data_size=2); tile_d0/tensor_d0/stride_d0 in 4B units.
// pad_int_code: LDS pad every 2^(code+1) DWORDs; pad_amt_code: (code+1) DWORDs.
__device__ inline void tdm_load_2d(const void* lds_dst, const void* gsrc,
                                   unsigned tile_d0, unsigned tile_d1,
                                   unsigned tensor_d0, unsigned tensor_d1,
                                   unsigned stride_d0,
                                   unsigned pad_int_code, unsigned pad_amt_code) {
  unsigned long long ga = (unsigned long long)gsrc;
  tdm_v4u g0;
  g0[0] = 1u;                                            // count=1, user mode
  g0[1] = (unsigned)(unsigned long long)lds_dst;         // lds_addr (bytes)
  g0[2] = (unsigned)ga;                                  // global_addr[31:0]
  g0[3] = ((unsigned)(ga >> 32) & 0x01FFFFFFu) | (2u << 30); // addr[56:32] | type=2
  tdm_v8i g1;
  g1[0] = (int)((2u << 16) | (1u << 20) |                // data_size=4B, pad_enable
                (pad_int_code << 22) | (pad_amt_code << 25));
  g1[1] = (int)((tensor_d0 & 0xFFFFu) << 16);            // tensor_dim0[15:0]
  g1[2] = (int)((tensor_d0 >> 16) | ((tensor_d1 & 0xFFFFu) << 16));
  g1[3] = (int)((tensor_d1 >> 16) | (tile_d0 << 16));    // tile_dim0
  g1[4] = (int)tile_d1;                                  // tile_dim1 (tile_dim2=0)
  g1[5] = (int)stride_d0;                                // tensor_dim0_stride[31:0]
  g1[6] = 0;
  g1[7] = 0;
  tdm_v4i z4 = {0, 0, 0, 0};
  tdm_v8i z8 = {0, 0, 0, 0, 0, 0, 0, 0};
  // 6-arg form (this toolchain): (g0, g1, g2, g3, g4, cpol)
  __builtin_amdgcn_tensor_load_to_lds(g0, g1, z4, z4, z8, 0);
}
__device__ inline void wait_tensor0() { __builtin_amdgcn_s_wait_tensorcnt(0); }
#else
#define HAVE_TDM 0
#endif

// ---------------------------------------------------------------------------
// fp32 -> bf16 bulk convert, 4 elements/thread (coalesced float4 in, u64 out)
// ---------------------------------------------------------------------------
__global__ __launch_bounds__(256)
void convert_f32_bf16_x4(const float* __restrict__ in,
                         bf16_t* __restrict__ out, long n4) {
  long i = (long)blockIdx.x * 256 + threadIdx.x;
  if (i >= n4) return;
  float4 f = ((const float4*)in)[i];
  unsigned long long p =
      (unsigned long long)bf_bits(f.x)
    | ((unsigned long long)bf_bits(f.y) << 16)
    | ((unsigned long long)bf_bits(f.z) << 32)
    | ((unsigned long long)bf_bits(f.w) << 48);
  ((unsigned long long*)out)[i] = p;
}

// ---------------------------------------------------------------------------
// Gating MLP: w[b,e] = (elu(elu(z[:,idx]@Wg1+bg1)@Wg2+bg2))@Wg3+bg3
// One block = 128 rows, one thread per row; z/g rows staged in LDS (padded).
// ---------------------------------------------------------------------------
#define GPAD 1
#define G_ROW (GH + GPAD)   // 129

__global__ __launch_bounds__(128)
void gating_kernel(const float* __restrict__ z,
                   const long long* __restrict__ idx,
                   const float* __restrict__ Wg1, const float* __restrict__ bg1,
                   const float* __restrict__ Wg2, const float* __restrict__ bg2,
                   const float* __restrict__ Wg3, const float* __restrict__ bg3,
                   float* __restrict__ gw) {
  extern __shared__ float gsm[];
  float* zbuf = gsm;               // [128][129]
  float* gbuf = gsm + 128 * G_ROW; // [128][129]

  const int t    = threadIdx.x;
  const int row0 = blockIdx.x * 128;
  const int col  = (int)idx[t];    // idx = arange(128)

  for (int j = 0; j < 128; ++j)
    zbuf[j * G_ROW + t] = z[(size_t)(row0 + j) * DIM + col];
  __syncthreads();

  for (int o = 0; o < GH; ++o) {
    float acc = bg1[o];
    for (int i = 0; i < GH; ++i) acc += zbuf[t * G_ROW + i] * Wg1[i * GH + o];
    gbuf[t * G_ROW + o] = elu_f(acc);
  }
  for (int o = 0; o < GH; ++o) {
    float acc = bg2[o];
    for (int i = 0; i < GH; ++i) acc += gbuf[t * G_ROW + i] * Wg2[i * GH + o];
    zbuf[t * G_ROW + o] = elu_f(acc);
  }
  for (int e = 0; e < NEXP; ++e) {
    float acc = bg3[e];
    for (int i = 0; i < GH; ++i) acc += zbuf[t * G_ROW + i] * Wg3[i * NEXP + e];
    gw[(size_t)(row0 + t) * NEXP + e] = acc;
  }
}
#define GATING_LDS_BYTES (2 * 128 * G_ROW * 4)

// ---------------------------------------------------------------------------
// Soft-MoE expert layer:  out[b,o] = act( sum_e w[b,e] * ((X@W_e)[b,o] + b_e[o]) )
// Block = 256 threads (8 waves, wave32), 128x128 output tile.
// A-strip (128x512 bf16) resident in LDS for all 8 experts (CDNA5 320KB LDS),
// staged with per-lane global_load_async_to_lds_b128 (ASYNCcnt).
// B tiles: BK=32, double-buffered, staged by the Tensor Data Mover (one
// tensor_load_to_lds per stage, TENSORcnt), TDM pad reproduces B_STRIDE.
// ---------------------------------------------------------------------------
#define APAD 8
#define BPAD 8
#define BK   32
#define A_STRIDE (DIM + APAD)        // 520 halves, 16B-aligned rows
#define B_STRIDE (128 + BPAD)        // 136 halves = 64+4 DWORDs, TDM pad layout
#define LDS_A_ELEMS (128 * A_STRIDE)
#define LDS_B_ELEMS (BK * B_STRIDE)  // one stage buffer
#define LAYER_LDS_BYTES ((LDS_A_ELEMS + 2 * LDS_B_ELEMS) * 2 + 128 * NEXP * 4)

// issue one 32x128-bf16 B-tile stage into `buf`
__device__ inline void stage_b_issue(const bf16_t* __restrict__ W, int e, int k0,
                                     int n0, bf16_t* buf, int tid, int wave) {
#if HAVE_TDM
  if (wave == 0)
    tdm_load_2d(buf, W + ((size_t)e * DIM + k0) * DIM + n0,
                /*tile_d0=*/128 / 2, /*tile_d1=*/BK,      // 64 x 32 (4B elems)
                /*tensor_d0=*/DIM / 2, /*tensor_d1=*/DIM, // 256 x 512
                /*stride_d0=*/DIM / 2,
                /*pad_int_code=*/5,   // pad every 64 DWORDs (one 128-col row)
                /*pad_amt_code=*/3);  // by 4 DWORDs  -> B_STRIDE = 136 halves
#else
#pragma unroll
  for (int j = 0; j < 2; ++j) {                 // 32x128 halves = 512 chunks / 256 thr
    int i = tid + j * 256;
    int r = i >> 4;
    int c = (i & 15) * 8;
    async_copy_b128(buf + r * B_STRIDE + c,
                    W + ((size_t)e * DIM + (k0 + r)) * DIM + n0 + c);
  }
#endif
}
__device__ inline void stage_b_wait(int wave) {
#if HAVE_TDM
  if (wave == 0) wait_tensor0();
#else
  wait_async0();
#endif
}

template <bool ACT, typename OUT_T>
__global__ __launch_bounds__(256)
void expert_layer_kernel(const bf16_t* __restrict__ X,     // [B, 512] bf16
                         const bf16_t* __restrict__ W,     // [E, 512, 512] bf16 (k-major)
                         const float*  __restrict__ bias,  // [E, 512] fp32
                         const float*  __restrict__ gw,    // [B, 8]   fp32
                         OUT_T*        __restrict__ out) { // [B, 512]
  extern __shared__ char smem[];
  bf16_t* As  = (bf16_t*)smem;                   // [128][A_STRIDE]
  bf16_t* Bs0 = As + LDS_A_ELEMS;                // [BK][B_STRIDE] stage buffer 0
  bf16_t* Bs1 = Bs0 + LDS_B_ELEMS;               // [BK][B_STRIDE] stage buffer 1
  float*  ws  = (float*)(Bs1 + LDS_B_ELEMS);     // [128][8] gate weights

  const int tid  = threadIdx.x;
  const int lane = tid & 31;
  const int wave = tid >> 5;
  const int wm   = wave & 3;       // wave row group: rows wm*32 .. +31
  const int wn   = wave >> 2;      // wave col group: cols wn*64 .. +63
  const int row0 = blockIdx.y * 128;
  const int n0   = blockIdx.x * 128;

  // ---- async-stage full A strip (128 x 512 bf16 = 128KB) into LDS ----
  for (int i = tid; i < 128 * (DIM / 8); i += 256) {
    int r = i >> 6;               // 64 chunks per row
    int c = (i & 63) * 8;
    async_copy_b128(As + r * A_STRIDE + c, X + (size_t)(row0 + r) * DIM + c);
  }
  for (int i = tid; i < 128 * NEXP; i += 256)
    ws[i] = gw[(size_t)row0 * NEXP + i];

  // prologue: expert 0, stage 0 into Bs0
  stage_b_issue(W, 0, 0, n0, Bs0, tid, wave);
  wait_async0();
  stage_b_wait(wave);
  __syncthreads();

  v8f oacc[2][4];
#pragma unroll
  for (int mt = 0; mt < 2; ++mt)
#pragma unroll
    for (int nt = 0; nt < 4; ++nt)
#pragma unroll
      for (int k = 0; k < 8; ++k) oacc[mt][nt][k] = 0.0f;

  const int mlane = lane & 15;
  const int khalf = (lane >> 4) * 8;   // A fragment K split by lane half

  for (int e = 0; e < NEXP; ++e) {
    v8f acc[2][4];
#pragma unroll
    for (int mt = 0; mt < 2; ++mt)
#pragma unroll
      for (int nt = 0; nt < 4; ++nt)
#pragma unroll
        for (int k = 0; k < 8; ++k) acc[mt][nt][k] = 0.0f;

    for (int s = 0; s < DIM / BK; ++s) {         // 16 K-stages of 32
      bf16_t* Bcur = (s & 1) ? Bs1 : Bs0;
      bf16_t* Bnxt = (s & 1) ? Bs0 : Bs1;
      // prefetch next stage (or next expert's first stage) while computing
      if (s + 1 < DIM / BK)   stage_b_issue(W, e,     (s + 1) * BK, n0, Bnxt, tid, wave);
      else if (e + 1 < NEXP)  stage_b_issue(W, e + 1, 0,            n0, Bnxt, tid, wave);

      const int k0 = s * BK;
      // A fragments: lane = row (m = lane&15), K {0..7,16..23}/{8..15,24..31}
      Frag a[2];
#pragma unroll
      for (int mt = 0; mt < 2; ++mt) {
        const bf16_t* p = As + (wm * 32 + mt * 16 + mlane) * A_STRIDE + k0;
        a[mt].u[0] = *(const uint4*)(p + khalf);
        a[mt].u[1] = *(const uint4*)(p + 16 + khalf);
      }
      // B fragments: lane = N column (n = lane&15), K {0..7,16..23}/{8..15,24..31}
      Frag b[4];
      const bf16_t* bp = Bcur + wn * 64 + mlane;
#pragma unroll
      for (int nt = 0; nt < 4; ++nt) {
#pragma unroll
        for (int i = 0; i < 16; ++i) {
          const int kk = (i < 8) ? (khalf + i) : (16 + khalf + (i - 8));
          b[nt].v[i] = bp[kk * B_STRIDE + nt * 16];
        }
      }
#pragma unroll
      for (int mt = 0; mt < 2; ++mt)
#pragma unroll
        for (int nt = 0; nt < 4; ++nt)
          acc[mt][nt] = __builtin_amdgcn_wmma_f32_16x16x32_bf16(
              false, a[mt].v, false, b[nt].v, (short)0, acc[mt][nt],
              false, false);

      stage_b_wait(wave);  // in-flight DMA for next stage has landed
      __syncthreads();     // visible to all; my reads finished pre-wmma
    }

    // fold per-row gate weight w[b,e] into the running output accumulator.
    // C/D layout: VGPR v, lanes 0-15 -> M=v, lanes 16-31 -> M=v+8.
#pragma unroll
    for (int mt = 0; mt < 2; ++mt) {
      const int rbase = wm * 32 + mt * 16 + (lane >> 4) * 8;
      float wv[8];
#pragma unroll
      for (int v = 0; v < 8; ++v) wv[v] = ws[(rbase + v) * NEXP + e];
#pragma unroll
      for (int nt = 0; nt < 4; ++nt)
#pragma unroll
        for (int v = 0; v < 8; ++v) oacc[mt][nt][v] += wv[v] * acc[mt][nt][v];
    }
  }

  // ---- epilogue: + sum_e w[b,e]*bias_e[o], activation, store ----
#pragma unroll
  for (int mt = 0; mt < 2; ++mt) {
    const int rbase = wm * 32 + mt * 16 + (lane >> 4) * 8;
    float wv2[8][NEXP];
#pragma unroll
    for (int v = 0; v < 8; ++v)
#pragma unroll
      for (int e = 0; e < NEXP; ++e) wv2[v][e] = ws[(rbase + v) * NEXP + e];
#pragma unroll
    for (int nt = 0; nt < 4; ++nt) {
      const int col = n0 + wn * 64 + nt * 16 + mlane;
      float bcol[NEXP];
#pragma unroll
      for (int e = 0; e < NEXP; ++e) bcol[e] = bias[e * DIM + col];
#pragma unroll
      for (int v = 0; v < 8; ++v) {
        float val = oacc[mt][nt][v];
#pragma unroll
        for (int e = 0; e < NEXP; ++e) val += wv2[v][e] * bcol[e];
        if (ACT) val = elu_f(val);
        store_val(out + (size_t)(row0 + rbase + v) * DIM + col, val);
      }
    }
  }
}

// ---------------------------------------------------------------------------
// Launcher
// ---------------------------------------------------------------------------
extern "C" void kernel_launch(void* const* d_in, const int* in_sizes, int n_in,
                              void* d_out, int out_size, void* d_ws, size_t ws_size,
                              hipStream_t stream) {
  const float*     z   = (const float*)d_in[0];
  const long long* idx = (const long long*)d_in[1];
  const float* Wg1 = (const float*)d_in[2];  const float* bg1 = (const float*)d_in[3];
  const float* Wg2 = (const float*)d_in[4];  const float* bg2 = (const float*)d_in[5];
  const float* Wg3 = (const float*)d_in[6];  const float* bg3 = (const float*)d_in[7];
  const float* W1  = (const float*)d_in[8];  const float* b1  = (const float*)d_in[9];
  const float* W2  = (const float*)d_in[10]; const float* b2  = (const float*)d_in[11];
  const float* W3  = (const float*)d_in[12]; const float* b3  = (const float*)d_in[13];
  float* out = (float*)d_out;

  // workspace layout (all 16B aligned)
  char* wsb = (char*)d_ws;
  const size_t act_bytes = (size_t)B_ROWS * DIM * sizeof(bf16_t);   // 64 MB
  bf16_t* zb  = (bf16_t*)wsb;                              // bf16 z (also reused as h2)
  bf16_t* h1  = (bf16_t*)(wsb + act_bytes);                // bf16 h1
  float*  gw  = (float*)(wsb + 2 * act_bytes);             // [B,8] gate logits
  bf16_t* Wt1 = (bf16_t*)(wsb + 2 * act_bytes + (size_t)B_ROWS * NEXP * 4);
  bf16_t* Wt2 = Wt1 + (size_t)NEXP * DIM * DIM;
  bf16_t* Wt3 = Wt2 + (size_t)NEXP * DIM * DIM;
  (void)in_sizes; (void)n_in; (void)out_size; (void)ws_size;

  // 1) fp32 -> bf16 conversions (activations + expert weights, layout kept)
  {
    long n4 = (long)B_ROWS * DIM / 4;
    convert_f32_bf16_x4<<<(unsigned)((n4 + 255) / 256), 256, 0, stream>>>(z, zb, n4);
    long w4 = (long)NEXP * DIM * DIM / 4;
    unsigned wg = (unsigned)((w4 + 255) / 256);
    convert_f32_bf16_x4<<<wg, 256, 0, stream>>>(W1, Wt1, w4);
    convert_f32_bf16_x4<<<wg, 256, 0, stream>>>(W2, Wt2, w4);
    convert_f32_bf16_x4<<<wg, 256, 0, stream>>>(W3, Wt3, w4);
  }

  // 2) gating MLP -> gw[B,8]
  gating_kernel<<<B_ROWS / 128, 128, GATING_LDS_BYTES, stream>>>(
      z, idx, Wg1, bg1, Wg2, bg2, Wg3, bg3, gw);

  // 3) three soft-MoE expert layers (WMMA bf16, fp32 accumulate,
  //    TDM + async-to-LDS staging)
  dim3 grid(DIM / 128, B_ROWS / 128);   // (4, 512)
  expert_layer_kernel<true,  bf16_t><<<grid, 256, LAYER_LDS_BYTES, stream>>>(
      zb, Wt1, b1, gw, h1);
  expert_layer_kernel<true,  bf16_t><<<grid, 256, LAYER_LDS_BYTES, stream>>>(
      h1, Wt2, b2, gw, zb /* h2 reuses zb */);
  expert_layer_kernel<false, float ><<<grid, 256, LAYER_LDS_BYTES, stream>>>(
      zb, Wt3, b3, gw, out);
}
